// DetGraphEmb_6871947673996
// MI455X (gfx1250) — matrix operand order repaired and checked by reading, not compiled, not run on hardware
//
#include <hip/hip_runtime.h>
#include <hip/hip_bf16.h>
#include <math.h>

typedef __attribute__((ext_vector_type(16))) _Float16 v16h;
typedef __attribute__((ext_vector_type(8)))  _Float16 v8h;
typedef __attribute__((ext_vector_type(8)))  float    v8f;
typedef __attribute__((ext_vector_type(4)))  int      v4i;

#define N_NODES 4096
#define E_EDGES 65536
#define FEA     128
#define NLAYERS 8
#define KSPLIT  4
#define LEAKY_SLOPE 0.01f

#if defined(__gfx1250__) && __has_builtin(__builtin_amdgcn_global_load_async_to_lds_b128) && __has_builtin(__builtin_amdgcn_s_wait_asynccnt)
#define USE_ASYNC_LDS 1
typedef __attribute__((address_space(1))) v4i gv4i;   // global int4
typedef __attribute__((address_space(3))) v4i lv4i;   // LDS int4
#else
#define USE_ASYNC_LDS 0
#endif

__device__ __forceinline__ float leakyf(float x) { return x > 0.f ? x : LEAKY_SLOPE * x; }

__device__ __forceinline__ v16h load_bfrag(const _Float16* bp) {
    // bp is 16-byte aligned; two ds_load_b128 -> one v16h fragment
    v8h lo = *(const v8h*)bp;
    v8h hi = *(const v8h*)(bp + 8);
    return __builtin_shufflevector(lo, hi, 0,1,2,3,4,5,6,7,8,9,10,11,12,13,14,15);
}

// ---------------------------------------------------------------------------
// Edge attention: a[e] = sigmoid( leaky(|h[s]-h[d]| @ W1 + b1) @ W2 + b2 )
// Block = 256 threads (8 waves), each wave owns 16 edges, full 128-wide hidden.
// ---------------------------------------------------------------------------
template <int KIN>
__global__ __launch_bounds__(256)
void edge_att_kernel(const float* __restrict__ h,
                     const int* __restrict__ src, const int* __restrict__ dst,
                     const float* __restrict__ W1, const float* __restrict__ b1,
                     const float* __restrict__ W2, const float* __restrict__ b2,
                     float* __restrict__ aout)
{
    constexpr int KP = (KIN + 31) & ~31;
    constexpr int KC = KP / 32;
    __shared__ _Float16 sW[FEA * KP];   // W1 transposed: sW[n*KP + k]
    __shared__ float sB1[FEA];
    __shared__ float sW2[FEA];

    const int tid = threadIdx.x;
    for (int i = tid; i < FEA * KP; i += 256) {
        int n = i / KP, k = i % KP;
        sW[i] = (k < KIN) ? (_Float16)W1[k * FEA + n] : (_Float16)0.0f;
    }
    if (tid < FEA) { sB1[tid] = b1[tid]; sW2[tid] = W2[tid]; }
    __syncthreads();

    const int lane  = tid & 31;
    const int wave  = tid >> 5;
    const int ebase = blockIdx.x * 128 + wave * 16;
    const int e     = ebase + (lane & 15);
    const int s     = src[e], d = dst[e];
    const int kb    = (lane < 16) ? 0 : 16;   // A-fragment K base per half-wave
    const int nb    = (lane < 16) ? 0 : 16;   // B-fragment K base per half-wave
    const int ncol  = lane & 15;

    v16h afrag[KC];
    #pragma unroll
    for (int c = 0; c < KC; ++c) {
        #pragma unroll
        for (int i = 0; i < 16; ++i) {
            int k = c * 32 + kb + i;
            float v = 0.f;
            if (k < KIN) v = fabsf(h[s * KIN + k] - h[d * KIN + k]);
            afrag[c][i] = (_Float16)v;
        }
    }

    float partial[8];
    #pragma unroll
    for (int i = 0; i < 8; ++i) partial[i] = 0.f;

    for (int t = 0; t < 8; ++t) {              // 8 column tiles of 16 -> 128 hidden
        const int n = t * 16 + ncol;
        v8f acc = {};
        #pragma unroll
        for (int c = 0; c < KC; ++c) {
            v16h bfrag = load_bfrag(&sW[n * KP + c * 32 + nb]);
            acc = __builtin_amdgcn_wmma_f32_16x16x32_f16(
                false, afrag[c], false, bfrag, (short)0, acc, false, false);
        }
        const float b1v = sB1[n];
        const float w2v = sW2[n];
        #pragma unroll
        for (int i = 0; i < 8; ++i)
            partial[i] += leakyf(acc[i] + b1v) * w2v;   // fused 128->1 layer
    }

    #pragma unroll
    for (int i = 0; i < 8; ++i) {               // reduce 16 columns per half-wave
        float p = partial[i];
        p += __shfl_xor(p, 1, 16);
        p += __shfl_xor(p, 2, 16);
        p += __shfl_xor(p, 4, 16);
        p += __shfl_xor(p, 8, 16);
        partial[i] = p;
    }
    const float b2v = b2[0];
    const int lr = lane & 15;
    if (lr < 8) {
        int row = lr + ((lane < 16) ? 0 : 8);
        float x = partial[lr] + b2v;
        aout[ebase + row] = 1.f / (1.f + __expf(-x));
    }
}

// ---------------------------------------------------------------------------
// Generic C[M x 128] = act(X[M x KIN] @ W[KIN x 128] + b)
// ---------------------------------------------------------------------------
template <int KIN>
__global__ __launch_bounds__(256)
void gemm_kernel(const float* __restrict__ X, const float* __restrict__ W,
                 const float* __restrict__ b, float* __restrict__ C, int do_leaky)
{
    constexpr int KP = (KIN + 31) & ~31;
    constexpr int KC = KP / 32;
    __shared__ _Float16 sW[FEA * KP];
    __shared__ float sB[FEA];

    const int tid = threadIdx.x;
    for (int i = tid; i < FEA * KP; i += 256) {
        int n = i / KP, k = i % KP;
        sW[i] = (k < KIN) ? (_Float16)W[k * FEA + n] : (_Float16)0.0f;
    }
    if (tid < FEA) sB[tid] = b[tid];
    __syncthreads();

    const int lane  = tid & 31;
    const int wave  = tid >> 5;
    const int rbase = blockIdx.x * 128 + wave * 16;
    const int row   = rbase + (lane & 15);
    const int kb    = (lane < 16) ? 0 : 16;
    const int nb    = (lane < 16) ? 0 : 16;
    const int ncol  = lane & 15;
    const int rhi   = (lane < 16) ? 0 : 8;

    v16h afrag[KC];
    #pragma unroll
    for (int c = 0; c < KC; ++c) {
        #pragma unroll
        for (int i = 0; i < 16; ++i) {
            int k = c * 32 + kb + i;
            afrag[c][i] = (k < KIN) ? (_Float16)X[row * KIN + k] : (_Float16)0.0f;
        }
    }

    for (int t = 0; t < 8; ++t) {
        const int n = t * 16 + ncol;
        v8f acc = {};
        #pragma unroll
        for (int c = 0; c < KC; ++c) {
            v16h bfrag = load_bfrag(&sW[n * KP + c * 32 + nb]);
            acc = __builtin_amdgcn_wmma_f32_16x16x32_f16(
                false, afrag[c], false, bfrag, (short)0, acc, false, false);
        }
        const float bias = sB[n];
        #pragma unroll
        for (int v = 0; v < 8; ++v) {
            float val = acc[v] + bias;
            if (do_leaky) val = leakyf(val);
            C[(rbase + v + rhi) * FEA + n] = val;
        }
    }
}

// ---------------------------------------------------------------------------
// Dense aggregation, split-K: Part[y][4096x128] = (uA*rinv)[:, Kslice] @ G[Kslice]
// Grid (32, KSPLIT); block = 128 rows x 128 cols; each y owns 1024 of K.
// G tile staged via async global->LDS (ASYNCcnt) when available.
// ---------------------------------------------------------------------------
__global__ __launch_bounds__(256)
void agg_kernel(const float* __restrict__ uA, const float* __restrict__ rinv,
                const float* __restrict__ G, float* __restrict__ part)
{
    __shared__ _Float16 sB[FEA * 32];   // G tile transposed: sB[n*32 + k]
#if USE_ASYNC_LDS
    __shared__ float sRaw[32 * FEA];    // raw f32 tile, filled by async DMA
#endif

    const int tid   = threadIdx.x;
    const int lane  = tid & 31;
    const int wave  = tid >> 5;
    const int rbase = blockIdx.x * 128 + wave * 16;
    const int arow  = rbase + (lane & 15);
    const float scale = rinv[arow];
    const int kb    = (lane < 16) ? 0 : 16;
    const int nb    = (lane < 16) ? 0 : 16;
    const int ncol  = lane & 15;
    const int rhi   = (lane < 16) ? 0 : 8;

    const int kcBeg = blockIdx.y * (N_NODES / 32 / KSPLIT);
    const int kcEnd = kcBeg + (N_NODES / 32 / KSPLIT);

    v8f acc[8];
    #pragma unroll
    for (int t = 0; t < 8; ++t) acc[t] = (v8f){};

    for (int kc = kcBeg; kc < kcEnd; ++kc) {
        __syncthreads();
#if USE_ASYNC_LDS
        {   // async DMA the contiguous 32x128 f32 tile of G into LDS (ASYNCcnt)
            const float* gp = G + (size_t)kc * 32 * FEA;
            #pragma unroll
            for (int c = 0; c < 4; ++c) {
                int idx = (tid + c * 256) * 4;          // 4 floats per b128
                __builtin_amdgcn_global_load_async_to_lds_b128(
                    (gv4i*)(float*)(gp + idx), (lv4i*)(sRaw + idx), 0, 0);
            }
            __builtin_amdgcn_s_wait_asynccnt(0);
            __syncthreads();
            for (int i = tid; i < 32 * FEA; i += 256) {   // f32 -> f16 transpose
                int k = i >> 7, n = i & 127;
                sB[n * 32 + k] = (_Float16)sRaw[i];
            }
        }
#else
        for (int i = tid; i < 32 * FEA; i += 256) {
            int k = i >> 7, n = i & 127;
            sB[n * 32 + k] = (_Float16)G[(kc * 32 + k) * FEA + n];
        }
#endif
        __syncthreads();

        if (kc + 8 < kcEnd)   // predictable uA stream: prefetch ~1KB ahead
            __builtin_prefetch(&uA[(size_t)arow * N_NODES + (kc + 8) * 32], 0, 1);

        v16h af;
        #pragma unroll
        for (int i = 0; i < 16; ++i) {
            int k = kc * 32 + kb + i;
            af[i] = (_Float16)(uA[(size_t)arow * N_NODES + k] * scale);
        }
        #pragma unroll
        for (int t = 0; t < 8; ++t) {
            v16h bfrag = load_bfrag(&sB[(t * 16 + ncol) * 32 + nb]);
            acc[t] = __builtin_amdgcn_wmma_f32_16x16x32_f16(
                false, af, false, bfrag, (short)0, acc[t], false, false);
        }
    }

    float* po = part + (size_t)blockIdx.y * N_NODES * FEA;
    #pragma unroll
    for (int t = 0; t < 8; ++t) {
        const int n = t * 16 + ncol;
        #pragma unroll
        for (int v = 0; v < 8; ++v)
            po[(rbase + v + rhi) * FEA + n] = acc[t][v];
    }
}

// fixed-order split-K reduction + leaky (deterministic)
__global__ __launch_bounds__(256)
void ksum_leaky_kernel(const float* __restrict__ part, float* __restrict__ out) {
    int i = blockIdx.x * 256 + threadIdx.x;        // over N_NODES*FEA
    float s = 0.f;
    #pragma unroll
    for (int p = 0; p < KSPLIT; ++p)
        s += part[(size_t)p * N_NODES * FEA + i];
    out[i] = leakyf(s);
}

// ---------------------------------------------------------------------------
// Sparse -> dense uA construction (deterministic: int atomicMax only)
// ---------------------------------------------------------------------------
__global__ void clear_u128_kernel(uint4* __restrict__ p, unsigned n4) {
    for (unsigned i = blockIdx.x * blockDim.x + threadIdx.x; i < n4;
         i += gridDim.x * blockDim.x)
        p[i] = make_uint4(0u, 0u, 0u, 0u);
}

__global__ void scatter_max_kernel(const int* __restrict__ src, const int* __restrict__ dst,
                                   unsigned* __restrict__ widx) {
    int e = blockIdx.x * blockDim.x + threadIdx.x;
    if (e < E_EDGES)
        atomicMax(&widx[(size_t)src[e] * N_NODES + dst[e]], (unsigned)(e + 1));
}

// converts winner-index buffer (in place) to uA float values; fixes diagonal
__global__ void fill_uA_kernel(float* __restrict__ uA, const float* __restrict__ a,
                               const float* __restrict__ A) {
    const unsigned n = (unsigned)N_NODES * (unsigned)N_NODES;
    for (unsigned i = blockIdx.x * blockDim.x + threadIdx.x; i < n;
         i += gridDim.x * blockDim.x) {
        unsigned v = ((const unsigned*)uA)[i];
        float ae = v ? a[v - 1] : 0.f;
        unsigned row = i >> 12, col = i & 4095u;
        float val;
        if (row == col)
            val = (A[i] > 0.f) ? 2.f * (1.f + ae) : 1.f;  // self-edge vs plain self-loop
        else
            val = ae;                                      // edge (A==1) or zero
        uA[i] = val;
    }
}

__global__ __launch_bounds__(256)
void rowsum_inv_kernel(const float* __restrict__ uA, float* __restrict__ rinv) {
    __shared__ float s[256];
    const int row = blockIdx.x;
    const float4* r = (const float4*)(uA + (size_t)row * N_NODES);
    float sum = 0.f;
    for (int c = threadIdx.x; c < N_NODES / 4; c += 256) {
        float4 v = r[c];
        sum += v.x + v.y + v.z + v.w;
    }
    s[threadIdx.x] = sum;
    __syncthreads();
    for (int st = 128; st > 0; st >>= 1) {
        if (threadIdx.x < st) s[threadIdx.x] += s[threadIdx.x + st];
        __syncthreads();
    }
    if (threadIdx.x == 0) rinv[row] = 1.f / s[0];
}

// ---------------------------------------------------------------------------
// Row L2 normalize (+ optional residual). One wave per row of 128.
// ---------------------------------------------------------------------------
__global__ __launch_bounds__(256)
void rownorm_kernel(const float* __restrict__ G, const float* __restrict__ prev,
                    float* __restrict__ out, int add_res) {
    const int row  = blockIdx.x * 8 + (threadIdx.x >> 5);
    const int lane = threadIdx.x & 31;
    float v[4];
    float ss = 0.f;
    #pragma unroll
    for (int i = 0; i < 4; ++i) {
        v[i] = G[row * FEA + lane + i * 32];
        ss += v[i] * v[i];
    }
    ss += __shfl_xor(ss, 1, 32);
    ss += __shfl_xor(ss, 2, 32);
    ss += __shfl_xor(ss, 4, 32);
    ss += __shfl_xor(ss, 8, 32);
    ss += __shfl_xor(ss, 16, 32);
    const float inv = 1.f / fmaxf(sqrtf(ss), 1e-12f);
    #pragma unroll
    for (int i = 0; i < 4; ++i) {
        float o = v[i] * inv;
        if (add_res) o += prev[row * FEA + lane + i * 32];
        out[row * FEA + lane + i * 32] = o;
    }
}

// ---------------------------------------------------------------------------
// Host orchestration
// ---------------------------------------------------------------------------
extern "C" void kernel_launch(void* const* d_in, const int* in_sizes, int n_in,
                              void* d_out, int out_size, void* d_ws, size_t ws_size,
                              hipStream_t stream) {
    (void)in_sizes; (void)n_in; (void)out_size; (void)ws_size;

    const float* x_box = (const float*)d_in[0];
    const float* A     = (const float*)d_in[1];
    const int*   eidx  = (const int*)d_in[2];
    const int*   src   = eidx;
    const int*   dst   = eidx + E_EDGES;

    // params: dict insertion order, each list layer-major
    const int BW1 = 3, BB1 = 11, BW2 = 19, BB2 = 27;
    const int AW1 = 35, AB1 = 43, AW2 = 51, AB2 = 59;

    float* uA   = (float*)d_ws;                          // 4096*4096
    float* aE   = uA + (size_t)N_NODES * N_NODES;        // 65536
    float* rinv = aE + E_EDGES;                          // 4096
    float* hA   = rinv + N_NODES;                        // 4096*128
    float* hB   = hA + N_NODES * FEA;
    float* tmid = hB + N_NODES * FEA;
    float* g1   = tmid + N_NODES * FEA;
    float* g2   = g1 + N_NODES * FEA;
    float* part = g2 + N_NODES * FEA;                    // KSPLIT * 4096*128

    const float* hcur = x_box;

    for (int l = 0; l < NLAYERS; ++l) {
        const float* aW1 = (const float*)d_in[AW1 + l];
        const float* ab1 = (const float*)d_in[AB1 + l];
        const float* aW2 = (const float*)d_in[AW2 + l];
        const float* ab2 = (const float*)d_in[AB2 + l];
        const float* bW1 = (const float*)d_in[BW1 + l];
        const float* bb1 = (const float*)d_in[BB1 + l];
        const float* bW2 = (const float*)d_in[BW2 + l];
        const float* bb2 = (const float*)d_in[BB2 + l];

        // 1) edge attention scores
        if (l == 0)
            edge_att_kernel<4><<<E_EDGES / 128, 256, 0, stream>>>(hcur, src, dst, aW1, ab1, aW2, ab2, aE);
        else
            edge_att_kernel<128><<<E_EDGES / 128, 256, 0, stream>>>(hcur, src, dst, aW1, ab1, aW2, ab2, aE);

        // 2) dense uA build: clear, winner scatter (last-write-wins), value fill + diag
        clear_u128_kernel<<<2048, 256, 0, stream>>>((uint4*)uA, (unsigned)(N_NODES * (size_t)N_NODES / 4));
        scatter_max_kernel<<<E_EDGES / 256, 256, 0, stream>>>(src, dst, (unsigned*)uA);
        fill_uA_kernel<<<4096, 256, 0, stream>>>(uA, aE, A);
        rowsum_inv_kernel<<<N_NODES, 256, 0, stream>>>(uA, rinv);

        // 3) node MLP
        if (l == 0)
            gemm_kernel<4><<<N_NODES / 128, 256, 0, stream>>>(hcur, bW1, bb1, tmid, 1);
        else
            gemm_kernel<128><<<N_NODES / 128, 256, 0, stream>>>(hcur, bW1, bb1, tmid, 1);
        gemm_kernel<128><<<N_NODES / 128, 256, 0, stream>>>(tmid, bW2, bb2, g1, 0);

        // 4) aggregation (split-K) + deterministic reduce + leaky
        agg_kernel<<<dim3(N_NODES / 128, KSPLIT), 256, 0, stream>>>(uA, rinv, g1, part);
        ksum_leaky_kernel<<<N_NODES * FEA / 256, 256, 0, stream>>>(part, g2);

        // 5) L2 row-normalize (+ residual for l=1..6); final layer writes d_out
        float* hnext = (l == NLAYERS - 1) ? (float*)d_out : ((l & 1) ? hB : hA);
        int add_res = (l >= 1 && l < NLAYERS - 1) ? 1 : 0;
        rownorm_kernel<<<N_NODES / 8, 256, 0, stream>>>(g2, hcur, hnext, add_res);

        hcur = hnext;
    }
}
